// Block_24532853194876
// MI455X (gfx1250) — compile-verified, matrix operand order checked by reading, not run on hardware
//
#include <hip/hip_runtime.h>
#include <hip/hip_bf16.h>
#include <cstdint>

// ---------------------------------------------------------------------------
// Types matching the CDNA5 WMMA builtins (probe-confirmed signatures)
// ---------------------------------------------------------------------------
typedef __attribute__((ext_vector_type(16))) __bf16 v16bf;
typedef __attribute__((ext_vector_type(8)))  float  v8f;

#define B_    64
#define NT    257      // tokens (16*16 window + 1 cls)
#define NKV   288      // keys padded to 9 chunks of 32
#define D_    1024
#define HD_   4096
#define NH    16
#define HDIM  64
#define MTOK  (B_*NT)  // 16448 = 257 * 64

// ---------------------------------------------------------------------------
// CDNA5 async global->LDS copy (ASYNCcnt path). Low 32 bits of a generic LDS
// pointer are the LDS byte address (ISA 10.2 aperture mapping).
// ---------------------------------------------------------------------------
__device__ __forceinline__ void async_copy16(const void* lds_dst, const void* gsrc) {
  unsigned loff = (unsigned)(uintptr_t)lds_dst;
  asm volatile("global_load_async_to_lds_b128 %0, %1, off"
               :: "v"(loff), "v"(gsrc) : "memory");
}
__device__ __forceinline__ void wait_async0() {
  asm volatile("s_wait_asynccnt 0" ::: "memory");
}

// A-matrix 16-bit fragment K pattern (ISA 7.12.2): vgpr v<4 -> kb+2v, else kb+16+2(v-4)
__device__ __forceinline__ int kpat(int v, int kb) {
  return (v < 4) ? (kb + 2 * v) : (kb + 16 + 2 * (v - 4));
}

// ---------------------------------------------------------------------------
// Weight cast f32 -> bf16
// ---------------------------------------------------------------------------
__global__ __launch_bounds__(256) void cast_bf16_kernel(const float* __restrict__ in,
                                                        __bf16* __restrict__ out, size_t n) {
  size_t i = (size_t)blockIdx.x * 256 + threadIdx.x;
  if (i < n) out[i] = (__bf16)in[i];
}

// ---------------------------------------------------------------------------
// LayerNorm over D (1024) -> bf16
// ---------------------------------------------------------------------------
__global__ __launch_bounds__(256) void ln_cast_kernel(const float* __restrict__ x,
                                                      const float* __restrict__ g,
                                                      const float* __restrict__ b,
                                                      __bf16* __restrict__ out, int D) {
  __shared__ float rs[8], rs2[8], stats[2];
  int row = blockIdx.x, tid = threadIdx.x;
  const float* xr = x + (size_t)row * D;
  float s = 0.f, s2 = 0.f;
  for (int i = tid; i < D; i += 256) { float v = xr[i]; s += v; s2 += v * v; }
  for (int o = 16; o; o >>= 1) { s += __shfl_xor(s, o, 32); s2 += __shfl_xor(s2, o, 32); }
  int w = tid >> 5, ln = tid & 31;
  if (!ln) { rs[w] = s; rs2[w] = s2; }
  __syncthreads();
  if (tid == 0) {
    float S = 0.f, S2 = 0.f;
    for (int i = 0; i < 8; ++i) { S += rs[i]; S2 += rs2[i]; }
    float mu = S / D, var = S2 / D - mu * mu;
    stats[0] = mu; stats[1] = rsqrtf(var + 1e-5f);
  }
  __syncthreads();
  float mu = stats[0], rstd = stats[1];
  __bf16* orow = out + (size_t)row * D;
  for (int i = tid; i < D; i += 256)
    orow[i] = (__bf16)((xr[i] - mu) * rstd * g[i] + b[i]);
}

// ---------------------------------------------------------------------------
// silu(h1)*h2 -> LayerNorm over HD (4096) -> bf16
// ---------------------------------------------------------------------------
__global__ __launch_bounds__(256) void swiglu_ln_kernel(const float* __restrict__ h1,
                                                        const float* __restrict__ h2,
                                                        const float* __restrict__ g,
                                                        const float* __restrict__ b,
                                                        __bf16* __restrict__ out) {
  __shared__ float sbuf[HD_];
  __shared__ float rs[8], rs2[8], stats[2];
  int row = blockIdx.x, tid = threadIdx.x;
  const float* a = h1 + (size_t)row * HD_;
  const float* c = h2 + (size_t)row * HD_;
  float s = 0.f, s2 = 0.f;
  for (int i = tid; i < HD_; i += 256) {
    float v1 = a[i], v2 = c[i];
    float sv = v1 / (1.f + __expf(-v1)) * v2;   // silu(v1)*v2
    sbuf[i] = sv; s += sv; s2 += sv * sv;
  }
  for (int o = 16; o; o >>= 1) { s += __shfl_xor(s, o, 32); s2 += __shfl_xor(s2, o, 32); }
  int w = tid >> 5, ln = tid & 31;
  if (!ln) { rs[w] = s; rs2[w] = s2; }
  __syncthreads();
  if (tid == 0) {
    float S = 0.f, S2 = 0.f;
    for (int i = 0; i < 8; ++i) { S += rs[i]; S2 += rs2[i]; }
    float mu = S / HD_, var = S2 / HD_ - mu * mu;
    stats[0] = mu; stats[1] = rsqrtf(var + 1e-5f);
  }
  __syncthreads();
  float mu = stats[0], rstd = stats[1];
  __bf16* orow = out + (size_t)row * HD_;
  for (int i = tid; i < HD_; i += 256)
    orow[i] = (__bf16)((sbuf[i] - mu) * rstd * g[i] + b[i]);
}

// ---------------------------------------------------------------------------
// Tiled bf16 WMMA GEMM:  C[M,N] = A[M,K] * W[N,K]^T  (+bias)(+resid)
// Block tile 64x128, K-step 32, 8 waves, each wave a 32x32 sub-tile
// (4 WMMAs with A/B fragment reuse).  Double-buffered LDS staged with the
// CDNA5 async global->LDS engine: next K-slab is in flight while the current
// one is consumed; ONE barrier per K-step.
// mode 0: f32 out (+bias,+resid)     mode 1: bf16 QKV scatter [B,H,NKV,64]*scale
// ---------------------------------------------------------------------------
__global__ __launch_bounds__(256) void gemm_bf16_kernel(const __bf16* __restrict__ A,
                                                        const __bf16* __restrict__ W,
                                                        const float* __restrict__ bias,
                                                        const float* __restrict__ resid,
                                                        float* __restrict__ outF,
                                                        __bf16* __restrict__ outQ,
                                                        int M, int N, int K,
                                                        int mode, float scale) {
  __shared__ __bf16 Al[2][64][40];    // stride 40 keeps 16B alignment
  __shared__ __bf16 Bl[2][128][40];
  int tid = threadIdx.x;
  int m0 = blockIdx.y * 64, n0 = blockIdx.x * 128;
  int w = tid >> 5, lane = tid & 31;
  int wm = (w & 1) * 32;              // wave m-offset in block tile
  int wn = (w >> 1) * 32;             // wave n-offset in block tile

  // staging map: A 64x32 -> 1x16B/thread, B 128x32 -> 2x16B/thread
  int a_row = tid >> 2, a_col = (tid & 3) * 8;
  int b_row = tid >> 1, b_col = (tid & 1) * 16;

  auto stage = [&](int bi, int kk0) {
    async_copy16(&Al[bi][a_row][a_col], A + (size_t)(m0 + a_row) * K + kk0 + a_col);
    async_copy16(&Bl[bi][b_row][b_col],     W + (size_t)(n0 + b_row) * K + kk0 + b_col);
    async_copy16(&Bl[bi][b_row][b_col + 8], W + (size_t)(n0 + b_row) * K + kk0 + b_col + 8);
  };

  v8f acc[4] = {};
  stage(0, 0);
  int buf = 0;
  int cr = lane & 15, kb8 = (lane >> 4) * 8;
  for (int k0 = 0; k0 < K; k0 += 32) {
    wait_async0();              // loads into `buf` complete (this wave)
    __syncthreads();            // all waves' loads visible; prev reads of buf^1 done
    if (k0 + 32 < K) stage(buf ^ 1, k0 + 32);   // prefetch next slab async

    // ---- fragments per ISA 7.12.2 layouts ----
    v16bf afr[2], bfr[2];
#pragma unroll
    for (int v = 0; v < 8; ++v) {
      int kk = kpat(v, kb8);
      afr[0][2 * v]     = Al[buf][wm + cr][kk];
      afr[0][2 * v + 1] = Al[buf][wm + cr][kk + 1];
      afr[1][2 * v]     = Al[buf][wm + 16 + cr][kk];
      afr[1][2 * v + 1] = Al[buf][wm + 16 + cr][kk + 1];
      bfr[0][2 * v]     = Bl[buf][wn + cr][kk];
      bfr[0][2 * v + 1] = Bl[buf][wn + cr][kk + 1];
      bfr[1][2 * v]     = Bl[buf][wn + 16 + cr][kk];
      bfr[1][2 * v + 1] = Bl[buf][wn + 16 + cr][kk + 1];
    }
    acc[0] = __builtin_amdgcn_wmma_f32_16x16x32_bf16(false, afr[0], false, bfr[0], (short)0, acc[0], false, false);
    acc[1] = __builtin_amdgcn_wmma_f32_16x16x32_bf16(false, afr[0], false, bfr[1], (short)0, acc[1], false, false);
    acc[2] = __builtin_amdgcn_wmma_f32_16x16x32_bf16(false, afr[1], false, bfr[0], (short)0, acc[2], false, false);
    acc[3] = __builtin_amdgcn_wmma_f32_16x16x32_bf16(false, afr[1], false, bfr[1], (short)0, acc[3], false, false);
    buf ^= 1;
  }

  // ---- epilogue (C layout: row = r + half*8, col = lane&15) ----
  int col = lane & 15, half = lane >> 4;
#pragma unroll
  for (int sa = 0; sa < 2; ++sa) {
#pragma unroll
    for (int sb = 0; sb < 2; ++sb) {
      const v8f& av = acc[sa * 2 + sb];
#pragma unroll
      for (int r = 0; r < 8; ++r) {
        int m = m0 + wm + sa * 16 + half * 8 + r;
        int n = n0 + wn + sb * 16 + col;
        if (m >= M) continue;
        float v = av[r] + (bias ? bias[n] : 0.f);
        if (mode == 0) {
          if (resid) v += resid[(size_t)m * N + n];
          outF[(size_t)m * N + n] = v;
        } else {
          v *= scale;
          int bb = m / NT, t = m % NT;
          int h = n >> 6, d = n & 63;
          outQ[(((size_t)(bb * NH + h)) * NKV + t) * HDIM + d] = (__bf16)v;
        }
      }
    }
  }
}

// ---------------------------------------------------------------------------
// RoPE on q and k, heads 1..15, tokens 1..256 (in place, bf16)
// ---------------------------------------------------------------------------
__global__ __launch_bounds__(256) void rope_kernel(__bf16* __restrict__ q, __bf16* __restrict__ k,
                                                   const float* __restrict__ cs,
                                                   const float* __restrict__ sn) {
  size_t idx = (size_t)blockIdx.x * 256 + threadIdx.x;
  const size_t total = (size_t)B_ * 15 * 256 * 32;
  if (idx >= total) return;
  int p = (int)(idx & 31); idx >>= 5;
  int t = (int)(idx & 255) + 1; idx >>= 8;
  int h = (int)(idx % 15) + 1;
  int b = (int)(idx / 15);
  size_t base = (((size_t)(b * NH + h)) * NKV + t) * HDIM + 2 * p;
  float c0 = cs[(t - 1) * 64 + 2 * p], c1 = cs[(t - 1) * 64 + 2 * p + 1];
  float s0 = sn[(t - 1) * 64 + 2 * p], s1 = sn[(t - 1) * 64 + 2 * p + 1];
  {
    float e = (float)q[base], o = (float)q[base + 1];
    q[base]     = (__bf16)(e * c0 - o * s0);
    q[base + 1] = (__bf16)(o * c1 + e * s1);
  }
  {
    float e = (float)k[base], o = (float)k[base + 1];
    k[base]     = (__bf16)(e * c0 - o * s0);
    k[base + 1] = (__bf16)(o * c1 + e * s1);
  }
}

// Relative-position bias index (matches REL_IDX construction)
__device__ __forceinline__ int rel_idx(int i, int j) {
  if (i == 0 && j == 0) return 963;
  if (i == 0) return 961;
  if (j == 0) return 962;
  int ih = (i - 1) >> 4, iw = (i - 1) & 15;
  int jh = (j - 1) >> 4, jw = (j - 1) & 15;
  return (ih - jh + 15) * 31 + (iw - jw + 15);
}

// ---------------------------------------------------------------------------
// Fused flash attention: one wave per (b,h,q-tile). Online softmax; WMMA for
// QK^T (K-dim = 64 channels, 2 chained wmmas) and PV (K-dim = 32 keys).
// Scores+bias never hit HBM.
// ---------------------------------------------------------------------------
__global__ __launch_bounds__(32) void attn_kernel(const __bf16* __restrict__ q,
                                                  const __bf16* __restrict__ k,
                                                  const __bf16* __restrict__ v,
                                                  const float* __restrict__ rel_tab,
                                                  float* __restrict__ out) {
  __shared__ __bf16 P[16][40];   // probability tile staged C-layout -> A-layout
  int qt = blockIdx.x % 17;
  int bh = blockIdx.x / 17;
  int h = bh % NH, b = bh / NH;
  int lane = threadIdx.x;
  int row = lane & 15, kb8 = (lane >> 4) * 8;
  int col = lane & 15, half = lane >> 4;

  const __bf16* qp = q + ((size_t)bh * NKV + qt * 16) * HDIM;
  const __bf16* kp = k + (size_t)bh * NKV * HDIM;
  const __bf16* vp = v + (size_t)bh * NKV * HDIM;

  // Q fragments for the two 32-wide d-chunks (A layout)
  v16bf qf[2];
#pragma unroll
  for (int c = 0; c < 2; ++c)
#pragma unroll
    for (int vv = 0; vv < 8; ++vv) {
      int kk = kpat(vv, kb8);
      qf[c][2 * vv]     = qp[row * HDIM + c * 32 + kk];
      qf[c][2 * vv + 1] = qp[row * HDIM + c * 32 + kk + 1];
    }

  float mrow[8], lrow[8];
  v8f O[4] = {};
#pragma unroll
  for (int r = 0; r < 8; ++r) { mrow[r] = -1e30f; lrow[r] = 0.f; }

  for (int kc = 0; kc < 9; ++kc) {     // 9 chunks x 32 keys = 288 (padded)
    // ---- S = Q K^T for two 16-key tiles ----
    v8f S[2];
#pragma unroll
    for (int t = 0; t < 2; ++t) {
      v8f acc = {};
#pragma unroll
      for (int c = 0; c < 2; ++c) {
        v16bf kf;
#pragma unroll
        for (int vv = 0; vv < 8; ++vv) {
          int kk = kpat(vv, kb8);
          size_t kr = (size_t)(kc * 32 + t * 16 + col) * HDIM + c * 32;
          kf[2 * vv]     = kp[kr + kk];
          kf[2 * vv + 1] = kp[kr + kk + 1];
        }
        acc = __builtin_amdgcn_wmma_f32_16x16x32_bf16(false, qf[c], false, kf, (short)0, acc, false, false);
      }
      S[t] = acc;
    }

    // ---- bias + mask + online softmax (row reductions across 16-lane half) ----
#pragma unroll
    for (int r = 0; r < 8; ++r) {
      int qg = qt * 16 + half * 8 + r;
#pragma unroll
      for (int t = 0; t < 2; ++t) {
        int kg = kc * 32 + t * 16 + col;
        float sv = S[t][r];
        if (qg < NT && kg < NT) sv += rel_tab[rel_idx(qg, kg) * NH + h];
        else                    sv = -1e30f;
        S[t][r] = sv;
      }
      float cm = fmaxf(S[0][r], S[1][r]);
      for (int o = 8; o; o >>= 1) cm = fmaxf(cm, __shfl_xor(cm, o, 16));
      float mnew = fmaxf(mrow[r], cm);
      float alpha = __expf(mrow[r] - mnew);
      mrow[r] = mnew;
      lrow[r] *= alpha;
#pragma unroll
      for (int dt = 0; dt < 4; ++dt) O[dt][r] *= alpha;
      float ps = 0.f;
#pragma unroll
      for (int t = 0; t < 2; ++t) {
        float p = __expf(S[t][r] - mnew);
        S[t][r] = p;
        ps += p;
      }
      for (int o = 8; o; o >>= 1) ps += __shfl_xor(ps, o, 16);
      lrow[r] += ps;
#pragma unroll
      for (int t = 0; t < 2; ++t)
        P[half * 8 + r][t * 16 + col] = (__bf16)S[t][r];
    }
    __syncthreads();

    // ---- O += P V (A-frag from LDS, V B-frags from global) ----
    v16bf pf;
#pragma unroll
    for (int vv = 0; vv < 8; ++vv) {
      int kk = kpat(vv, kb8);
      pf[2 * vv]     = P[row][kk];
      pf[2 * vv + 1] = P[row][kk + 1];
    }
#pragma unroll
    for (int dt = 0; dt < 4; ++dt) {
      v16bf vf;
#pragma unroll
      for (int vv = 0; vv < 8; ++vv) {
        int kk = kpat(vv, kb8);
        vf[2 * vv]     = vp[(size_t)(kc * 32 + kk) * HDIM + dt * 16 + col];
        vf[2 * vv + 1] = vp[(size_t)(kc * 32 + kk + 1) * HDIM + dt * 16 + col];
      }
      O[dt] = __builtin_amdgcn_wmma_f32_16x16x32_bf16(false, pf, false, vf, (short)0, O[dt], false, false);
    }
    __syncthreads();
  }

  // ---- normalize + store [b, q, h*64+d] ----
#pragma unroll
  for (int r = 0; r < 8; ++r) {
    int qg = qt * 16 + half * 8 + r;
    if (qg >= NT) continue;
    float inv = 1.f / lrow[r];
#pragma unroll
    for (int dt = 0; dt < 4; ++dt)
      out[((size_t)b * NT + qg) * D_ + h * HDIM + dt * 16 + col] = O[dt][r] * inv;
  }
}

// ---------------------------------------------------------------------------
// Host-side orchestration
// ---------------------------------------------------------------------------
extern "C" void kernel_launch(void* const* d_in, const int* in_sizes, int n_in,
                              void* d_out, int out_size, void* d_ws, size_t ws_size,
                              hipStream_t stream) {
  (void)in_sizes; (void)n_in; (void)out_size; (void)ws_size;
  const float* x        = (const float*)d_in[0];
  const float* rope_cos = (const float*)d_in[1];
  const float* rope_sin = (const float*)d_in[2];
  const float* q_w  = (const float*)d_in[3];
  const float* q_b  = (const float*)d_in[4];
  const float* k_w  = (const float*)d_in[5];
  const float* v_w  = (const float*)d_in[6];
  const float* v_b  = (const float*)d_in[7];
  const float* rel_tab = (const float*)d_in[8];
  const float* iln_g = (const float*)d_in[9];
  const float* iln_b = (const float*)d_in[10];
  const float* proj_w = (const float*)d_in[11];
  const float* proj_b = (const float*)d_in[12];
  const float* n1_g = (const float*)d_in[13];
  const float* n1_b = (const float*)d_in[14];
  const float* n2_g = (const float*)d_in[15];
  const float* n2_b = (const float*)d_in[16];
  const float* w1_w = (const float*)d_in[17];
  const float* w1_b = (const float*)d_in[18];
  const float* w2_w = (const float*)d_in[19];
  const float* w2_b = (const float*)d_in[20];
  const float* fln_g = (const float*)d_in[21];
  const float* fln_b = (const float*)d_in[22];
  const float* w3_w = (const float*)d_in[23];
  const float* w3_b = (const float*)d_in[24];
  float* out = (float*)d_out;

  char* ws = (char*)d_ws;
  size_t off = 0;
  auto alloc = [&](size_t bytes) -> char* {
    char* p = ws + off;
    off += (bytes + 255) & ~(size_t)255;
    return p;
  };
  const size_t M = MTOK;
  __bf16* qw_bf = (__bf16*)alloc((size_t)D_ * D_ * 2);
  __bf16* kw_bf = (__bf16*)alloc((size_t)D_ * D_ * 2);
  __bf16* vw_bf = (__bf16*)alloc((size_t)D_ * D_ * 2);
  __bf16* pw_bf = (__bf16*)alloc((size_t)D_ * D_ * 2);
  __bf16* w1_bf = (__bf16*)alloc((size_t)HD_ * D_ * 2);
  __bf16* w2_bf = (__bf16*)alloc((size_t)HD_ * D_ * 2);
  __bf16* w3_bf = (__bf16*)alloc((size_t)D_ * HD_ * 2);
  __bf16* xn_bf = (__bf16*)alloc(M * D_ * 2);
  __bf16* qbuf  = (__bf16*)alloc((size_t)B_ * NH * NKV * HDIM * 2);
  __bf16* kbuf  = (__bf16*)alloc((size_t)B_ * NH * NKV * HDIM * 2);
  __bf16* vbuf  = (__bf16*)alloc((size_t)B_ * NH * NKV * HDIM * 2);
  float*  attn_out = (float*)alloc(M * D_ * 4);
  __bf16* y_bf  = (__bf16*)alloc(M * D_ * 2);
  float*  xmid  = (float*)alloc(M * D_ * 4);
  __bf16* xn2_bf = (__bf16*)alloc(M * D_ * 2);
  float*  h1    = (float*)alloc(M * HD_ * 4);
  float*  h2    = (float*)alloc(M * HD_ * 4);
  __bf16* hn_bf = (__bf16*)alloc(M * HD_ * 2);

  auto cast = [&](const float* src, __bf16* dst, size_t n) {
    cast_bf16_kernel<<<(unsigned)((n + 255) / 256), 256, 0, stream>>>(src, dst, n);
  };
  cast(q_w, qw_bf, (size_t)D_ * D_);
  cast(k_w, kw_bf, (size_t)D_ * D_);
  cast(v_w, vw_bf, (size_t)D_ * D_);
  cast(proj_w, pw_bf, (size_t)D_ * D_);
  cast(w1_w, w1_bf, (size_t)HD_ * D_);
  cast(w2_w, w2_bf, (size_t)HD_ * D_);
  cast(w3_w, w3_bf, (size_t)D_ * HD_);

  // LN1 -> bf16
  ln_cast_kernel<<<(unsigned)M, 256, 0, stream>>>(x, n1_g, n1_b, xn_bf, D_);

  // QKV projections (WMMA GEMM), scattered to [B,H,NKV,64] bf16; q pre-scaled
  dim3 gqkv(D_ / 128, (unsigned)(M / 64));
  gemm_bf16_kernel<<<gqkv, 256, 0, stream>>>(xn_bf, qw_bf, q_b, nullptr, nullptr, qbuf,
                                             (int)M, D_, D_, 1, 0.125f);
  gemm_bf16_kernel<<<gqkv, 256, 0, stream>>>(xn_bf, kw_bf, nullptr, nullptr, nullptr, kbuf,
                                             (int)M, D_, D_, 1, 1.0f);
  gemm_bf16_kernel<<<gqkv, 256, 0, stream>>>(xn_bf, vw_bf, v_b, nullptr, nullptr, vbuf,
                                             (int)M, D_, D_, 1, 1.0f);

  // RoPE on q,k (heads 1..15, tokens 1..256)
  {
    size_t total = (size_t)B_ * 15 * 256 * 32;
    rope_kernel<<<(unsigned)((total + 255) / 256), 256, 0, stream>>>(qbuf, kbuf, rope_cos, rope_sin);
  }

  // Fused attention with relative-position bias and online softmax
  attn_kernel<<<(unsigned)(B_ * NH * 17), 32, 0, stream>>>(qbuf, kbuf, vbuf, rel_tab, attn_out);

  // inner LN -> bf16, proj GEMM + bias + residual(x) -> xmid (f32)
  ln_cast_kernel<<<(unsigned)M, 256, 0, stream>>>(attn_out, iln_g, iln_b, y_bf, D_);
  gemm_bf16_kernel<<<gqkv, 256, 0, stream>>>(y_bf, pw_bf, proj_b, x, xmid, nullptr,
                                             (int)M, D_, D_, 0, 1.0f);

  // LN2 -> bf16, FFN up-projections
  ln_cast_kernel<<<(unsigned)M, 256, 0, stream>>>(xmid, n2_g, n2_b, xn2_bf, D_);
  dim3 gup(HD_ / 128, (unsigned)(M / 64));
  gemm_bf16_kernel<<<gup, 256, 0, stream>>>(xn2_bf, w1_bf, w1_b, nullptr, h1, nullptr,
                                            (int)M, HD_, D_, 0, 1.0f);
  gemm_bf16_kernel<<<gup, 256, 0, stream>>>(xn2_bf, w2_bf, w2_b, nullptr, h2, nullptr,
                                            (int)M, HD_, D_, 0, 1.0f);

  // silu(h1)*h2 -> LN(ffn) -> bf16, down-projection + bias + residual(xmid) -> out
  swiglu_ln_kernel<<<(unsigned)M, 256, 0, stream>>>(h1, h2, fln_g, fln_b, hn_bf);
  gemm_bf16_kernel<<<gqkv, 256, 0, stream>>>(hn_bf, w3_bf, w3_b, xmid, out, nullptr,
                                             (int)M, D_, HD_, 0, 1.0f);
}
